// LSTM_40492951667219
// MI455X (gfx1250) — matrix operand order repaired and checked by reading, not compiled, not run on hardware
//
#include <hip/hip_runtime.h>

// ---------------- problem dims ----------------
#define BS     32
#define SL     128
#define XS     512
#define HS     512
#define VOCAB  32000
#define ZDIM   (XS + HS)     // 1024
#define GCOLS  (4 * HS)      // 2048 (f|i|c|o)
#define NH     (BS * HS)     // 16384

typedef __attribute__((ext_vector_type(16))) __bf16 bf16x16;
typedef __attribute__((ext_vector_type(8)))  float  f32x8;
typedef unsigned short u16;

// ---------------- workspace layout (bytes) ----------------
static constexpr size_t CTR_OFF   = 0;                                   // barrier counter
static constexpr size_t W4T_OFF   = 256;                                 // bf16 [2048][1024]
static constexpr size_t W4T_BYTES = (size_t)GCOLS * ZDIM * 2;
static constexpr size_t B4_OFF    = W4T_OFF + W4T_BYTES;                 // f32  [2048]
static constexpr size_t B4_BYTES  = (size_t)GCOLS * 4;
static constexpr size_t WVT_OFF   = B4_OFF + B4_BYTES;                   // bf16 [32000][512]
static constexpr size_t WVT_BYTES = (size_t)VOCAB * HS * 2;
static constexpr size_t XBF_OFF   = WVT_OFF + WVT_BYTES;                 // bf16 [SL][BS][XS]
static constexpr size_t XBF_BYTES = (size_t)SL * BS * XS * 2;
static constexpr size_t HALL_OFF  = XBF_OFF + XBF_BYTES;                 // bf16 [SL][BS][HS]
static constexpr size_t HALL_BYTES= (size_t)SL * BS * HS * 2;
static constexpr size_t H0_OFF    = HALL_OFF + HALL_BYTES;               // bf16 [BS][HS]
static constexpr size_t H0_BYTES  = (size_t)NH * 2;
static constexpr size_t GATES_OFF = H0_OFF + H0_BYTES;                   // f32  [BS][2048]

// LDS panel strides (elements): pad rows by 8 elems (16B) -> row stride = 16 mod 256 bytes
// => the 16 lanes of a B-fragment ds_load_b128 hit 16 distinct 16B chunks = all 64 banks.
#define RSTRIDE_REC  (ZDIM + 8)   // 1032 elems = 2064 B per row, 128 rows = 264192 B
#define RSTRIDE_PROJ (HS + 8)     // 520 elems  = 1040 B per row, 64 rows  = 66560 B

// ---------------- helpers ----------------
__device__ __forceinline__ u16 f2bf(float f) {
  unsigned u = __float_as_uint(f);
  u += 0x7FFFu + ((u >> 16) & 1u);      // round-to-nearest-even
  return (u16)(u >> 16);
}
__device__ __forceinline__ float sigmoidf_(float x) { return 1.0f / (1.0f + __expf(-x)); }

union BFrag { uint4 q[2]; bf16x16 v; };

// A fragment: 16x32 bf16, row = lane%16; lane-half h takes K = {16h..16h+15}
__device__ __forceinline__ bf16x16 load_a_frag(const u16* __restrict__ rowp, int half) {
  BFrag f;
  f.q[0] = *(const uint4*)(rowp + half * 16);
  f.q[1] = *(const uint4*)(rowp + half * 16 + 8);
  return f.v;
}
// B fragment from any memory (N-major): per-lane contiguous 32B along K
__device__ __forceinline__ bf16x16 load_b_frag(const u16* __restrict__ p) {
  BFrag f;
  f.q[0] = *(const uint4*)(p);
  f.q[1] = *(const uint4*)(p + 8);
  return f.v;
}

// low 32 bits of a generic LDS pointer == wave-relative LDS byte address (ISA 10.2)
__device__ __forceinline__ unsigned lds_off(const void* p) { return (unsigned)(size_t)p; }

// CDNA5 async copy: global -> LDS, 16B per lane, tracked by ASYNCcnt
__device__ __forceinline__ void async_g2l_b128(unsigned laddr, const void* gaddr) {
  asm volatile("global_load_async_to_lds_b128 %0, %1, off" :: "v"(laddr), "v"(gaddr) : "memory");
}
__device__ __forceinline__ void wait_async0() {
  asm volatile("s_wait_asynccnt 0x0" ::: "memory");
}

// agent-scope grid barrier (monotonic counter)
__device__ __forceinline__ void gbar(unsigned* c, unsigned target) {
  __syncthreads();
  __threadfence();                       // release my stores to agent scope
  if (threadIdx.x == 0) {
    __hip_atomic_fetch_add(c, 1u, __ATOMIC_ACQ_REL, __HIP_MEMORY_SCOPE_AGENT);
    while (__hip_atomic_load(c, __ATOMIC_RELAXED, __HIP_MEMORY_SCOPE_AGENT) < target)
      __builtin_amdgcn_s_sleep(2);
  }
  __syncthreads();
  __threadfence();                       // acquire: drop stale cached lines
}

// ---------------- setup kernels ----------------
__global__ void __launch_bounds__(256) k_init(const float* __restrict__ h_pre,
                                              const float* __restrict__ C_pre,
                                              u16* __restrict__ h0,
                                              float* __restrict__ Cst,
                                              unsigned* __restrict__ ctr) {
  int idx = blockIdx.x * 256 + threadIdx.x;
  if (idx == 0) *ctr = 0u;
  if (idx < NH) { h0[idx] = f2bf(h_pre[idx]); Cst[idx] = C_pre[idx]; }
}

__global__ void __launch_bounds__(256) k_bias(const float* __restrict__ bf_, const float* __restrict__ bi_,
                                              const float* __restrict__ bc_, const float* __restrict__ bo_,
                                              float* __restrict__ bias4) {
  int n = blockIdx.x * 256 + threadIdx.x;
  if (n < GCOLS) {
    int g = n >> 9;
    const float* B = (g == 0) ? bf_ : (g == 1) ? bi_ : (g == 2) ? bc_ : bo_;
    bias4[n] = B[n & (HS - 1)];
  }
}

// pack wf|wi|wc|wo [1024,512] f32 -> W4t [2048][1024] bf16 (N-major), LDS tiled transpose
__global__ void __launch_bounds__(256) k_pack_w(const float* __restrict__ wf, const float* __restrict__ wi,
                                                const float* __restrict__ wc, const float* __restrict__ wo,
                                                u16* __restrict__ w4t) {
  __shared__ float tile[32][33];
  int n0 = blockIdx.x * 32, k0 = blockIdx.y * 32;
  int tx = threadIdx.x & 31, ty = threadIdx.x >> 5;
  int g = n0 >> 9;                                   // whole tile inside one gate (512 % 32 == 0)
  const float* W = (g == 0) ? wf : (g == 1) ? wi : (g == 2) ? wc : wo;
  int nn0 = n0 & (HS - 1);
#pragma unroll
  for (int r = 0; r < 4; ++r)
    tile[ty + 8 * r][tx] = W[(size_t)(k0 + ty + 8 * r) * HS + nn0 + tx];
  __syncthreads();
#pragma unroll
  for (int r = 0; r < 4; ++r)
    w4t[(size_t)(n0 + ty + 8 * r) * ZDIM + k0 + tx] = f2bf(tile[tx][ty + 8 * r]);
}

// wv [512,32000] f32 -> wvt [32000][512] bf16 (one-shot read: non-temporal)
__global__ void __launch_bounds__(256) k_tr_wv(const float* __restrict__ wv, u16* __restrict__ wvt) {
  __shared__ float tile[32][33];
  int n0 = blockIdx.x * 32, k0 = blockIdx.y * 32;
  int tx = threadIdx.x & 31, ty = threadIdx.x >> 5;
#pragma unroll
  for (int r = 0; r < 4; ++r)
    tile[ty + 8 * r][tx] = __builtin_nontemporal_load(&wv[(size_t)(k0 + ty + 8 * r) * VOCAB + n0 + tx]);
  __syncthreads();
#pragma unroll
  for (int r = 0; r < 4; ++r)
    wvt[(size_t)(n0 + ty + 8 * r) * HS + k0 + tx] = f2bf(tile[tx][ty + 8 * r]);
}

// gather + convert: x_bf[s][b][k] = bf16(embedding[input[b][s]][k])
__global__ void __launch_bounds__(256) k_embed(const int* __restrict__ inp, const float* __restrict__ emb,
                                               u16* __restrict__ xbf) {
  size_t idx = (size_t)blockIdx.x * 256 + threadIdx.x;   // < SL*BS*XS
  int k = (int)(idx & (XS - 1));
  int b = (int)((idx >> 9) & (BS - 1));
  int s = (int)(idx >> 14);
  int row = inp[b * SL + s];
  xbf[idx] = f2bf(emb[(size_t)row * XS + k]);
}

// ---------------- persistent recurrent kernel: 32 WGs x 8 waves = 256 waves = 256 tiles ----------
// Per-WG B panel (8 N-tiles x 1024 K, 256KB+pad) is async-staged into LDS ONCE and reused for
// all 128 timesteps; only the tiny z activations stream from L2 inside the serial loop.
__global__ void __launch_bounds__(256) k_lstm_recur(
    const u16* __restrict__ xbf, const u16* __restrict__ w4t, const float* __restrict__ bias4,
    const u16* __restrict__ h0, u16* __restrict__ hall, float* __restrict__ gates,
    float* __restrict__ Cst, float* __restrict__ hlast, unsigned* __restrict__ ctr) {
  extern __shared__ u16 bpanel[];                            // [128][RSTRIDE_REC]
  const int tid  = threadIdx.x;
  const int lane = tid & 31;
  const int half = lane >> 4;
  const int ln16 = lane & 15;
  const int wave = (blockIdx.x * 256 + tid) >> 5;            // 0..255
  const int mt = wave >> 7;                                  // 0..1   (batch tile)
  const int nt = wave & 127;                                 // 0..127 (gate-col tile)
  const int w  = tid >> 5;                                   // wave in block 0..7
  const int gtid = blockIdx.x * 256 + tid;
  const int nthreads = gridDim.x * 256;
  unsigned barcnt = 0;

  // ---- async-stage this WG's 128 B-rows of w4t into LDS (16B chunks, 64 per thread) ----
  {
    const int row0 = (blockIdx.x & 15) * 128;                // first w4t row for this WG
    const unsigned lbase = lds_off(bpanel);
    const u16* gbase = w4t + (size_t)row0 * ZDIM;
#pragma unroll 4
    for (int c = tid; c < 128 * (ZDIM / 8); c += 256) {      // 16384 chunks
      int row = c >> 7, seg = c & 127;
      async_g2l_b128(lbase + (unsigned)(row * (RSTRIDE_REC * 2) + seg * 16),
                     gbase + (size_t)row * ZDIM + seg * 8);
    }
    wait_async0();
    __syncthreads();
  }

  const int arow = 16 * mt + ln16;                           // batch row this lane loads for A
  const u16* brow = bpanel + (size_t)(w * 16 + ln16) * RSTRIDE_REC;  // LDS B row for this lane
  const float bias = bias4[16 * nt + ln16];

  for (int t = 0; t < SL; ++t) {
    const u16* xrow = xbf + ((size_t)t * BS + arow) * XS;
    const u16* hrow = ((t == 0) ? h0 : (hall + (size_t)(t - 1) * NH)) + (size_t)arow * HS;

    f32x8 acc = {0.f, 0.f, 0.f, 0.f, 0.f, 0.f, 0.f, 0.f};
#pragma unroll 4
    for (int kk = 0; kk < 16; ++kk) {                        // K = 0..511  (x part of z)
      bf16x16 a = load_a_frag(xrow + kk * 32, half);
      bf16x16 b = load_b_frag(brow + kk * 32 + half * 16);   // ds_load_b128 x2
      acc = __builtin_amdgcn_wmma_f32_16x16x32_bf16(false, a, false, b, (short)0, acc, false, false);
    }
#pragma unroll 4
    for (int kk = 0; kk < 16; ++kk) {                        // K = 512..1023 (h part of z)
      bf16x16 a = load_a_frag(hrow + kk * 32, half);
      bf16x16 b = load_b_frag(brow + (16 + kk) * 32 + half * 16);
      acc = __builtin_amdgcn_wmma_f32_16x16x32_bf16(false, a, false, b, (short)0, acc, false, false);
    }
    // scatter C/D tile: element i -> (M = i + 8*half, N = ln16)
#pragma unroll
    for (int i = 0; i < 8; ++i) {
      int r = 16 * mt + i + 8 * half;
      gates[(size_t)r * GCOLS + 16 * nt + ln16] = acc[i] + bias;
    }
    gbar(ctr, (++barcnt) * gridDim.x);

    // elementwise gates -> C, h
    for (int idx = gtid; idx < NH; idx += nthreads) {
      int b = idx >> 9, j = idx & (HS - 1);
      const float* g = gates + (size_t)b * GCOLS;
      float fg = sigmoidf_(g[j]);
      float ig = sigmoidf_(g[HS + j]);
      float cg = tanhf(g[2 * HS + j]);
      float og = sigmoidf_(g[3 * HS + j]);
      float C  = fg * Cst[idx] + ig * cg;
      float h  = og * tanhf(C);
      Cst[idx] = C;                    // running cell state == d_out C_last region
      hlast[idx] = h;                  // d_out h_last region (final step persists)
      hall[(size_t)t * NH + idx] = f2bf(h);
    }
    gbar(ctr, (++barcnt) * gridDim.x);
  }
}

// ---------------- projection: [4096,512] x [512,32000] + bv ------------------------------------
// All 8 waves of a block share one 64-column B panel (async-staged, 64KB); NT stores keep the
// 524MB out_prob stream from evicting L2-resident wvt/hall.
__global__ void __launch_bounds__(256) k_proj(const u16* __restrict__ hall, const u16* __restrict__ wvt,
                                              const float* __restrict__ bv, float* __restrict__ out) {
  extern __shared__ u16 bpanel[];                            // [64][RSTRIDE_PROJ]
  const int tid  = threadIdx.x;
  const int lane = tid & 31;
  const int half = lane >> 4;
  const int ln16 = lane & 15;
  const int g = blockIdx.x >> 5;                             // N-group 0..499 (64 cols), same for block
  const int m = (blockIdx.x & 31) * 8 + (tid >> 5);          // Mtile 0..255

  // ---- async-stage B panel: 64 rows x 512 K bf16 (16B chunks, 16 per thread) ----
  {
    const unsigned lbase = lds_off(bpanel);
    const u16* gbase = wvt + (size_t)(64 * g) * HS;
#pragma unroll 4
    for (int c = tid; c < 64 * (HS / 8); c += 256) {         // 4096 chunks
      int row = c >> 6, seg = c & 63;
      async_g2l_b128(lbase + (unsigned)(row * (RSTRIDE_PROJ * 2) + seg * 16),
                     gbase + (size_t)row * HS + seg * 8);
    }
    wait_async0();
    __syncthreads();
  }

  const u16* arow = hall + (size_t)(16 * m + ln16) * HS;

  f32x8 acc[4];
#pragma unroll
  for (int j = 0; j < 4; ++j) acc[j] = (f32x8){0.f, 0.f, 0.f, 0.f, 0.f, 0.f, 0.f, 0.f};

#pragma unroll 2
  for (int kk = 0; kk < 16; ++kk) {                          // K = 512
    bf16x16 a = load_a_frag(arow + kk * 32, half);
    __builtin_prefetch(arow + (kk + 1) * 32, 0, 0);          // global_prefetch_b8
#pragma unroll
    for (int j = 0; j < 4; ++j) {
      bf16x16 b = load_b_frag(bpanel + (size_t)(16 * j + ln16) * RSTRIDE_PROJ + kk * 32 + half * 16);
      acc[j] = __builtin_amdgcn_wmma_f32_16x16x32_bf16(false, a, false, b, (short)0, acc[j], false, false);
    }
  }
#pragma unroll
  for (int j = 0; j < 4; ++j) {
    int col = 64 * g + 16 * j + ln16;
    float bias = bv[col];
#pragma unroll
    for (int i = 0; i < 8; ++i) {
      int row = 16 * m + i + 8 * half;                       // == t*32 + batch
      __builtin_nontemporal_store(acc[j][i] + bias, &out[(size_t)row * VOCAB + col]);
    }
  }
}

// ---------------- host entry ----------------
extern "C" void kernel_launch(void* const* d_in, const int* in_sizes, int n_in,
                              void* d_out, int out_size, void* d_ws, size_t ws_size,
                              hipStream_t stream) {
  (void)in_sizes; (void)n_in; (void)out_size; (void)ws_size;
  const int*   inp  = (const int*)d_in[0];
  const float* emb  = (const float*)d_in[1];
  const float* wf   = (const float*)d_in[2];
  const float* wi   = (const float*)d_in[3];
  const float* wc   = (const float*)d_in[4];
  const float* wo   = (const float*)d_in[5];
  const float* wv   = (const float*)d_in[6];
  const float* bf_  = (const float*)d_in[7];
  const float* bi_  = (const float*)d_in[8];
  const float* bc_  = (const float*)d_in[9];
  const float* bo_  = (const float*)d_in[10];
  const float* bv   = (const float*)d_in[11];
  const float* h_pre = (const float*)d_in[12];
  const float* C_pre = (const float*)d_in[13];

  char* ws = (char*)d_ws;
  unsigned* ctr   = (unsigned*)(ws + CTR_OFF);
  u16*   w4t      = (u16*)(ws + W4T_OFF);
  float* bias4    = (float*)(ws + B4_OFF);
  u16*   wvt      = (u16*)(ws + WVT_OFF);
  u16*   xbf      = (u16*)(ws + XBF_OFF);
  u16*   hall     = (u16*)(ws + HALL_OFF);
  u16*   h0       = (u16*)(ws + H0_OFF);
  float* gates    = (float*)(ws + GATES_OFF);

  float* out   = (float*)d_out;
  float* hlast = out + (size_t)SL * BS * VOCAB;   // [32,512]
  float* Cst   = hlast + NH;                      // [32,512] (also running cell state)

  k_init <<<64, 256, 0, stream>>>(h_pre, C_pre, h0, Cst, ctr);
  k_bias <<<8, 256, 0, stream>>>(bf_, bi_, bc_, bo_, bias4);
  k_pack_w<<<dim3(GCOLS / 32, ZDIM / 32), 256, 0, stream>>>(wf, wi, wc, wo, w4t);
  k_tr_wv<<<dim3(VOCAB / 32, HS / 32), 256, 0, stream>>>(wv, wvt);
  k_embed<<<(SL * BS * XS) / 256, 256, 0, stream>>>(inp, emb, xbf);

  k_lstm_recur<<<32, 256, 128 * RSTRIDE_REC * 2, stream>>>(xbf, w4t, bias4, h0, hall, gates, Cst, hlast, ctr);
  k_proj<<<16000, 256, 64 * RSTRIDE_PROJ * 2, stream>>>(hall, wvt, bv, out);
}
